// MultiHeadSelfAttention_6545530159219
// MI455X (gfx1250) — compile-verified, matrix operand order checked by reading, not run on hardware
//
#include <hip/hip_runtime.h>
#include <hip/hip_bf16.h>
#include <math.h>

typedef __attribute__((ext_vector_type(16))) _Float16 v16h;
typedef __attribute__((ext_vector_type(8)))  _Float16 v8h;
typedef __attribute__((ext_vector_type(4)))  _Float16 v4h;
typedef __attribute__((ext_vector_type(8)))  float    v8f;
typedef __attribute__((ext_vector_type(4)))  float    v4f;

#define DIMD   512
#define QKVN   1536
#define NHEAD  8
#define HD     64
#define SEQ    2048
#define BATCH  4
#define NTOK   (BATCH * SEQ)   // 8192

// WMMA 16-bit operand fragment (A- and B-operand share the per-lane pattern).
// Lane l, row = l%16. halfs[0..7]  = K (l<16?0:8)+0..7
//                     halfs[8..15] = K 16+(l<16?0:8)+0..7
// 'off' must be (lane<16 ? 0 : 8).
__device__ __forceinline__ v16h make_frag(const _Float16* p, int off) {
    v8h lo = *(const v8h*)(p + off);
    v8h hi = *(const v8h*)(p + 16 + off);
    v16h f;
#pragma unroll
    for (int i = 0; i < 8; ++i) { f[i] = lo[i]; f[i + 8] = hi[i]; }
    return f;
}

__device__ __forceinline__ v8f wmma_f16(v16h a, v16h b, v8f c) {
    return __builtin_amdgcn_wmma_f32_16x16x32_f16(
        false, a, false, b, (short)0, c, false, false);
}

// ---------------------------------------------------------------------------
// Kernel 1: QKV = x @ w_qkv + b_qkv -> f16 Q[B,H,N,hd] (pre-scaled 1/8),
// K[B,H,N,hd], V transposed Vt[B,H,hd,N].
// 256 threads (8 waves), 128x128 tile, K-step 32. Section routing is uniform
// per block (nblk is 128-aligned, sections 512-wide) -> scalar branch only.
// ---------------------------------------------------------------------------
__global__ __launch_bounds__(256) void qkv_kernel(
    const float* __restrict__ x, const float* __restrict__ w,
    const float* __restrict__ bias,
    _Float16* __restrict__ qw, _Float16* __restrict__ kw,
    _Float16* __restrict__ vtw)
{
    __shared__ _Float16 As[128][32];
    __shared__ _Float16 Bs[128][32];

    const int tid  = threadIdx.x;
    const int lane = tid & 31;
    const int wave = tid >> 5;
    const int mblk = blockIdx.y * 128;
    const int nblk = blockIdx.x * 128;
    const int wm = (wave >> 2) * 64;
    const int wn = (wave & 3) * 32;
    const int half_off = (lane < 16) ? 0 : 8;
    const int rl = lane & 15;

    v8f acc[4][2] = {};

    for (int k0 = 0; k0 < DIMD; k0 += 32) {
        // stage A: x[mblk..+127][k0..+31] -> f16, packed 8B stores
        {
            int r = tid >> 3, c = (tid & 7) * 4;
#pragma unroll
            for (int it = 0; it < 4; ++it) {
                v4f xv = *(const v4f*)(x + (size_t)(mblk + r + it * 32) * DIMD + k0 + c);
                v4h hv;
#pragma unroll
                for (int j = 0; j < 4; ++j) hv[j] = (_Float16)xv[j];
                *(v4h*)&As[r + it * 32][c] = hv;
            }
        }
        // stage B transposed: thread owns one column n, 16 k's -> 2x b128 LDS
        {
            int n = tid & 127, kg = (tid >> 7) * 16;
            const float* wp = w + (size_t)(k0 + kg) * QKVN + nblk + n;
            v8h b8a, b8b;
#pragma unroll
            for (int t = 0; t < 8; ++t) b8a[t] = (_Float16)wp[(size_t)t * QKVN];
#pragma unroll
            for (int t = 0; t < 8; ++t) b8b[t] = (_Float16)wp[(size_t)(t + 8) * QKVN];
            *(v8h*)&Bs[n][kg]     = b8a;
            *(v8h*)&Bs[n][kg + 8] = b8b;
        }
        __syncthreads();

        v16h bf[2];
#pragma unroll
        for (int j = 0; j < 2; ++j) bf[j] = make_frag(&Bs[wn + j * 16 + rl][0], half_off);
#pragma unroll
        for (int i = 0; i < 4; ++i) {
            v16h af = make_frag(&As[wm + i * 16 + rl][0], half_off);
#pragma unroll
            for (int j = 0; j < 2; ++j) acc[i][j] = wmma_f16(af, bf[j], acc[i][j]);
        }
        __syncthreads();
    }

    const int sec = nblk >> 9;   // uniform per block: 0=Q 1=K 2=V
    if (sec == 2) {
        // V: 8 accum rows per lane are 8 consecutive n -> one b128 store each
#pragma unroll
        for (int i = 0; i < 4; ++i) {
#pragma unroll
            for (int j = 0; j < 2; ++j) {
                const int cc = (nblk & 511) + wn + j * 16;   // 16-aligned
                const int h  = cc >> 6;
                const int d  = (cc & 63) + rl;
                const float bv = bias[nblk + wn + j * 16 + rl];
                const int gr0 = mblk + wm + i * 16 + half_off;  // 8-aligned
                const int b   = gr0 >> 11;
                const int n0  = gr0 & 2047;
                v8h pk;
#pragma unroll
                for (int v = 0; v < 8; ++v) pk[v] = (_Float16)(acc[i][j][v] + bv);
                *(v8h*)&vtw[((size_t)(b * NHEAD + h) * HD + d) * SEQ + n0] = pk;
            }
        }
    } else {
        _Float16* __restrict__ dst = sec ? kw : qw;
        const float mul = sec ? 1.0f : 0.125f;
#pragma unroll
        for (int i = 0; i < 4; ++i) {
#pragma unroll
            for (int j = 0; j < 2; ++j) {
                const int cc = (nblk & 511) + wn + j * 16;
                const int h  = cc >> 6;
                const int d  = (cc & 63) + rl;
                const float bv = bias[nblk + wn + j * 16 + rl];
#pragma unroll
                for (int v = 0; v < 8; ++v) {
                    const int gr = mblk + wm + i * 16 + half_off + v;
                    const int b  = gr >> 11, n = gr & 2047;
                    dst[((size_t)(b * NHEAD + h) * SEQ + n) * HD + d] =
                        (_Float16)((acc[i][j][v] + bv) * mul);
                }
            }
        }
    }
}

// ---------------------------------------------------------------------------
// Kernel 2: flash attention, 32 queries per wave (2 query tiles share the
// K/V fragments). S^T = K_tile x Q; its C-layout (lane=query, vgpr=key) is
// the B-operand layout for O^T += V^T_tile x P after exp(). Two
// shfl_xor(16) per query-tile per 32-key block for online-softmax stats.
// ---------------------------------------------------------------------------
__global__ __launch_bounds__(256) void attn_kernel(
    const _Float16* __restrict__ qw, const _Float16* __restrict__ kw,
    const _Float16* __restrict__ vtw, _Float16* __restrict__ ow)
{
    const int tid  = threadIdx.x;
    const int lane = tid & 31;
    const int wave = tid >> 5;
    const int bh   = blockIdx.y;                       // 0..31
    const int half_off = (lane < 16) ? 0 : 8;
    const int rl = lane & 15;
    const int qrow0 = blockIdx.x * 256 + wave * 32 + rl;   // query tile 0
    // query tile 1 = qrow0 + 16

    v16h qf[2][2];
#pragma unroll
    for (int qt = 0; qt < 2; ++qt) {
        const _Float16* qptr = qw + ((size_t)bh * SEQ + qrow0 + qt * 16) * HD;
        qf[qt][0] = make_frag(qptr,      half_off);
        qf[qt][1] = make_frag(qptr + 32, half_off);
    }

    const _Float16* kbp = kw  + (size_t)bh * SEQ * HD;
    const _Float16* vbp = vtw + (size_t)bh * HD * SEQ;

    v8f ot[2][4] = {};
    float m_run[2] = {-INFINITY, -INFINITY};
    float l_run[2] = {0.f, 0.f};

    for (int kb = 0; kb < SEQ; kb += 32) {
        const _Float16* kp0 = kbp + (size_t)(kb + rl) * HD;
        const _Float16* kp1 = kbp + (size_t)(kb + 16 + rl) * HD;
        __builtin_prefetch(kp0 + 32 * HD, 0, 0);

        const v16h ka0 = make_frag(kp0,      half_off);
        const v16h ka1 = make_frag(kp0 + 32, half_off);
        const v16h kc0 = make_frag(kp1,      half_off);
        const v16h kc1 = make_frag(kp1 + 32, half_off);

        v16h pf[2];
#pragma unroll
        for (int qt = 0; qt < 2; ++qt) {
            v8f st0 = {};
            st0 = wmma_f16(ka0, qf[qt][0], st0);
            st0 = wmma_f16(ka1, qf[qt][1], st0);
            v8f st1 = {};
            st1 = wmma_f16(kc0, qf[qt][0], st1);
            st1 = wmma_f16(kc1, qf[qt][1], st1);

            float mloc = -INFINITY;
#pragma unroll
            for (int v = 0; v < 8; ++v) {
                mloc = fmaxf(mloc, st0[v]);
                mloc = fmaxf(mloc, st1[v]);
            }
            mloc = fmaxf(mloc, __shfl_xor(mloc, 16, 32));
            const float m_new = fmaxf(m_run[qt], mloc);
            const float scale = __expf(m_run[qt] - m_new);

            float p0[8], p1[8], sloc = 0.f;
#pragma unroll
            for (int v = 0; v < 8; ++v) {
                p0[v] = __expf(st0[v] - m_new);
                p1[v] = __expf(st1[v] - m_new);
                sloc += p0[v] + p1[v];
            }
            sloc += __shfl_xor(sloc, 16, 32);
            l_run[qt] = l_run[qt] * scale + sloc;
            m_run[qt] = m_new;

#pragma unroll
            for (int v = 0; v < 8; ++v) {
                pf[qt][v]     = (_Float16)p0[v];
                pf[qt][v + 8] = (_Float16)p1[v];
            }
#pragma unroll
            for (int dt = 0; dt < 4; ++dt)
#pragma unroll
                for (int v = 0; v < 8; ++v) ot[qt][dt][v] *= scale;
        }

#pragma unroll
        for (int dt = 0; dt < 4; ++dt) {
            const _Float16* vp = vbp + (size_t)(dt * 16 + rl) * SEQ + kb;
            const v16h vf = make_frag(vp, half_off);
            ot[0][dt] = wmma_f16(vf, pf[0], ot[0][dt]);
            ot[1][dt] = wmma_f16(vf, pf[1], ot[1][dt]);
        }
    }

    const int b = bh >> 3, h = bh & 7;
#pragma unroll
    for (int qt = 0; qt < 2; ++qt) {
        const float inv = 1.0f / l_run[qt];
        _Float16* orow =
            ow + ((size_t)(b * SEQ + qrow0 + qt * 16)) * DIMD + h * HD;
#pragma unroll
        for (int dt = 0; dt < 4; ++dt) {
            v8h o8;
#pragma unroll
            for (int v = 0; v < 8; ++v) o8[v] = (_Float16)(ot[qt][dt][v] * inv);
            *(v8h*)(orow + dt * 16 + half_off) = o8;
        }
    }
}

// ---------------------------------------------------------------------------
// Kernel 3: out = O @ w_out + b_out (f16 A, f32 out). 128x128 tile.
// ---------------------------------------------------------------------------
__global__ __launch_bounds__(256) void proj_kernel(
    const _Float16* __restrict__ ow, const float* __restrict__ w,
    const float* __restrict__ bias, float* __restrict__ out)
{
    __shared__ _Float16 As[128][32];
    __shared__ _Float16 Bs[128][32];

    const int tid  = threadIdx.x;
    const int lane = tid & 31;
    const int wave = tid >> 5;
    const int mblk = blockIdx.y * 128;
    const int nblk = blockIdx.x * 128;
    const int wm = (wave >> 2) * 64;
    const int wn = (wave & 3) * 32;
    const int half_off = (lane < 16) ? 0 : 8;
    const int rl = lane & 15;

    v8f acc[4][2] = {};

    for (int k0 = 0; k0 < DIMD; k0 += 32) {
        {   // stage A (f16 already): 16B copies
            int r = tid >> 2, c = (tid & 3) * 8;
#pragma unroll
            for (int it = 0; it < 2; ++it)
                *(v8h*)&As[r + it * 64][c] =
                    *(const v8h*)(ow + (size_t)(mblk + r + it * 64) * DIMD + k0 + c);
        }
        {   // stage B transposed, column-per-thread
            int n = tid & 127, kg = (tid >> 7) * 16;
            const float* wp = w + (size_t)(k0 + kg) * DIMD + nblk + n;
            v8h b8a, b8b;
#pragma unroll
            for (int t = 0; t < 8; ++t) b8a[t] = (_Float16)wp[(size_t)t * DIMD];
#pragma unroll
            for (int t = 0; t < 8; ++t) b8b[t] = (_Float16)wp[(size_t)(t + 8) * DIMD];
            *(v8h*)&Bs[n][kg]     = b8a;
            *(v8h*)&Bs[n][kg + 8] = b8b;
        }
        __syncthreads();

        v16h bf[2];
#pragma unroll
        for (int j = 0; j < 2; ++j) bf[j] = make_frag(&Bs[wn + j * 16 + rl][0], half_off);
#pragma unroll
        for (int i = 0; i < 4; ++i) {
            v16h af = make_frag(&As[wm + i * 16 + rl][0], half_off);
#pragma unroll
            for (int j = 0; j < 2; ++j) acc[i][j] = wmma_f16(af, bf[j], acc[i][j]);
        }
        __syncthreads();
    }

#pragma unroll
    for (int i = 0; i < 4; ++i) {
#pragma unroll
        for (int j = 0; j < 2; ++j) {
            const int c = nblk + wn + j * 16 + rl;
            const float bv = bias[c];
#pragma unroll
            for (int v = 0; v < 8; ++v) {
                const int gr = mblk + wm + i * 16 + half_off + v;
                out[(size_t)gr * DIMD + c] = acc[i][j][v] + bv;
            }
        }
    }
}

extern "C" void kernel_launch(void* const* d_in, const int* in_sizes, int n_in,
                              void* d_out, int out_size, void* d_ws, size_t ws_size,
                              hipStream_t stream) {
    const float* x     = (const float*)d_in[0];
    const float* w_qkv = (const float*)d_in[1];
    const float* b_qkv = (const float*)d_in[2];
    const float* w_out = (const float*)d_in[3];
    const float* b_out = (const float*)d_in[4];
    float* out = (float*)d_out;

    const size_t HELEMS = (size_t)BATCH * NHEAD * SEQ * HD;  // 4,194,304
    _Float16* qw = (_Float16*)d_ws;
    _Float16* kw = qw + HELEMS;
    _Float16* vt = kw + HELEMS;
    _Float16* ow = vt + HELEMS;   // [B*N, 512] f16  (32 MB total)

    qkv_kernel<<<dim3(QKVN / 128, NTOK / 128), 256, 0, stream>>>(
        x, w_qkv, b_qkv, qw, kw, vt);
    attn_kernel<<<dim3(SEQ / 256, BATCH * NHEAD), 256, 0, stream>>>(
        qw, kw, vt, ow);
    proj_kernel<<<dim3(DIMD / 128, NTOK / 128), 256, 0, stream>>>(
        ow, w_out, b_out, out);
}
